// _LinearBlockTransformation_73151882986149
// MI455X (gfx1250) — matrix operand order repaired and checked by reading, not compiled, not run on hardware
//
#include <hip/hip_runtime.h>
#include <hip/hip_bf16.h>
#include <stdint.h>

// Problem constants (match reference)
#define DIMB   256
#define BATCH  2048
#define IN_F   2048
#define OUT_F  2048
#define KSTEPS (IN_F / 32)    // 64  K-chunks of 32 for bf16 WMMA
#define MTILES (BATCH / 16)   // 128
#define NTILES (OUT_F / 16)   // 128

typedef __attribute__((ext_vector_type(16))) __bf16          v16bf;
typedef __attribute__((ext_vector_type(8)))  float           v8f;
typedef __attribute__((ext_vector_type(16))) unsigned short  v16us;
typedef __attribute__((ext_vector_type(8)))  unsigned short  v8us;

__device__ __forceinline__ unsigned short f32_to_bf16_rn(float f) {
  unsigned int u = __float_as_uint(f);
  unsigned int r = u + 0x7FFFu + ((u >> 16) & 1u);
  return (unsigned short)(r >> 16);
}
__device__ __forceinline__ float bf16u_to_f32(unsigned short h) {
  return __uint_as_float(((unsigned int)h) << 16);
}

// ---------------------------------------------------------------------------
// K1: per-row L2 norm of masked weight, write w_eff as bf16 hi/lo directly in
// WMMA B-fragment layout: frag (ks,nt) -> lane L holds B[ks*32+(L/16)*16+e,
// nt*16+L%16] for e=0..15, stored contiguously (32B/lane).
// One block (256 thr) per row r (= K index of B).
// ---------------------------------------------------------------------------
__global__ __launch_bounds__(256) void prep_b_kernel(
    const float* __restrict__ weight, const float* __restrict__ log_diag,
    float* __restrict__ log_norm, unsigned short* __restrict__ Bh,
    unsigned short* __restrict__ Bl) {
  const int r   = blockIdx.x;      // 0..2047
  const int tid = threadIdx.x;     // 0..255
  const int ri  = r >> 3;          // input block index
  __shared__ float red[256];

  const float4* wrow = (const float4*)(weight + (size_t)r * OUT_F);
  float4 wa = wrow[tid * 2 + 0];
  float4 wb = wrow[tid * 2 + 1];
  float wv[8] = {wa.x, wa.y, wa.z, wa.w, wb.x, wb.y, wb.z, wb.w};

  float ss = 0.f;
#pragma unroll
  for (int u = 0; u < 8; ++u) {
    int c  = tid * 8 + u;
    int cj = c >> 3;
    float val = (cj == ri) ? __expf(wv[u]) : ((ri < cj) ? wv[u] : 0.f);
    wv[u] = val;
    ss += val * val;
  }
  red[tid] = ss;
  __syncthreads();
#pragma unroll
  for (int s = 128; s > 0; s >>= 1) {
    if (tid < s) red[tid] += red[tid + s];
    __syncthreads();
  }
  float norm = sqrtf(red[0]);
  if (tid == 0) log_norm[r] = __logf(norm);
  float inv = 1.f / norm;

  const int ksIdx    = r >> 5;
  const int kin      = r & 31;
  const int laneHalf = (kin >> 4) << 4;  // 0 or 16
  const int e        = kin & 15;
#pragma unroll
  for (int u = 0; u < 8; ++u) {
    int c = tid * 8 + u;
    float we = __expf(log_diag[c]) * wv[u] * inv;
    unsigned short hi = f32_to_bf16_rn(we);
    float lof = we - bf16u_to_f32(hi);
    unsigned short lo = f32_to_bf16_rn(lof);
    int nt = c >> 4;
    int lane = laneHalf + (c & 15);
    size_t idx = (((size_t)ksIdx * NTILES + nt) * 32 + lane) * 16 + e;
    Bh[idx] = hi;
    Bl[idx] = lo;
  }
}

// ---------------------------------------------------------------------------
// K2: repack x (f32) -> bf16 hi/lo A-fragment layout.
// A frag (mt,ks): lane L holds x[mt*16+L%16, ks*32 + K0+e] for e=0..15
// with K0=(L/16)*16.  Thread = (mt,ks,lane,h): one contiguous 8-elem K run.
// ---------------------------------------------------------------------------
__global__ __launch_bounds__(256) void prep_a_kernel(
    const float* __restrict__ x, unsigned short* __restrict__ Ah,
    unsigned short* __restrict__ Al) {
  size_t t = (size_t)blockIdx.x * 256 + threadIdx.x;   // < 524288
  int h    = (int)(t & 1);
  int lane = (int)((t >> 1) & 31);
  int ks   = (int)((t >> 6) & 63);
  int mt   = (int)(t >> 12);

  int m  = mt * 16 + (lane & 15);
  int k0 = ks * 32 + ((lane >> 4) << 4) + h * 8;
  const float4* src = (const float4*)(x + (size_t)m * IN_F + k0);
  float4 a = src[0];
  float4 b = src[1];
  float v[8] = {a.x, a.y, a.z, a.w, b.x, b.y, b.z, b.w};

  v8us hv, lv;
#pragma unroll
  for (int i = 0; i < 8; ++i) {
    unsigned short hi = f32_to_bf16_rn(v[i]);
    float lof = v[i] - bf16u_to_f32(hi);
    hv[i] = hi;
    lv[i] = f32_to_bf16_rn(lof);
  }
  size_t dst = (((size_t)mt * KSTEPS + ks) * 32 + lane) * 16 + h * 8;
  *(v8us*)(Ah + dst) = hv;
  *(v8us*)(Al + dst) = lv;
}

// ---------------------------------------------------------------------------
// K3: ldb[d,i,j] = log_diag[8d+j] + weight[8d+i,8d+j] - log_norm[8d+i];
// my[d,j] = max_i ldb; eyb = exp(ldb - my).  Grid 256, block 64.
// ---------------------------------------------------------------------------
__global__ __launch_bounds__(64) void prep_ldb_kernel(
    const float* __restrict__ weight, const float* __restrict__ log_diag,
    const float* __restrict__ log_norm, float* __restrict__ eyb,
    float* __restrict__ my) {
  int d = blockIdx.x, tid = threadIdx.x;
  int i = tid >> 3, j = tid & 7;
  int r = d * 8 + i, c = d * 8 + j;
  float ldb = log_diag[c] + weight[(size_t)r * OUT_F + c] - log_norm[r];
  __shared__ float sm[64];
  sm[tid] = ldb;
  __syncthreads();
  float m = sm[j];
#pragma unroll
  for (int i2 = 1; i2 < 8; ++i2) m = fmaxf(m, sm[i2 * 8 + j]);
  if (i == 0) my[d * 8 + j] = m;
  eyb[d * 64 + i * 8 + j] = __expf(ldb - m);
}

// ---------------------------------------------------------------------------
// K4: GEMM y = x @ w_eff + bias via bf16x3 WMMA (hi*hi + lo*hi + hi*lo).
// 8 waves/block; wave -> 2 row tiles x 4 col tiles (32x64 output).
// Per K-step: 24 b128 loads feed 24 WMMAs (1.0 loads/WMMA). Grid (32,8).
// ---------------------------------------------------------------------------
__global__ __launch_bounds__(256) void gemm_bf16x3_kernel(
    const unsigned short* __restrict__ Ah, const unsigned short* __restrict__ Al,
    const unsigned short* __restrict__ Bh, const unsigned short* __restrict__ Bl,
    const float* __restrict__ bias, float* __restrict__ y) {
  const int lane = threadIdx.x & 31;
  const int wave = threadIdx.x >> 5;
  const int mt0 = blockIdx.y * 16 + wave * 2;  // 0..126 (pair mt0, mt0+1)
  const int nt0 = blockIdx.x * 4;              // 0..124

  const v8f vzero = {0.f, 0.f, 0.f, 0.f, 0.f, 0.f, 0.f, 0.f};
  v8f acc0[4], acc1[4];
#pragma unroll
  for (int j = 0; j < 4; ++j) { acc0[j] = vzero; acc1[j] = vzero; }

  const size_t ASTRIDE_MT = (size_t)KSTEPS * 512;  // elems between M-tiles
  size_t aoff  = ((size_t)mt0 * KSTEPS * 32 + lane) * 16;
  size_t boff0 = ((size_t)nt0 * 32 + lane) * 16;

  for (int ks = 0; ks < KSTEPS; ++ks) {
    v16bf a0h = __builtin_bit_cast(v16bf, *(const v16us*)(Ah + aoff));
    v16bf a0l = __builtin_bit_cast(v16bf, *(const v16us*)(Al + aoff));
    v16bf a1h = __builtin_bit_cast(v16bf, *(const v16us*)(Ah + aoff + ASTRIDE_MT));
    v16bf a1l = __builtin_bit_cast(v16bf, *(const v16us*)(Al + aoff + ASTRIDE_MT));
    __builtin_prefetch(Ah + aoff + 512, 0, 1);
    size_t boff = boff0 + (size_t)ks * NTILES * 512;
#pragma unroll
    for (int j = 0; j < 4; ++j) {
      v16bf b_h = __builtin_bit_cast(v16bf, *(const v16us*)(Bh + boff + (size_t)j * 512));
      v16bf b_l = __builtin_bit_cast(v16bf, *(const v16us*)(Bl + boff + (size_t)j * 512));
      acc0[j] = __builtin_amdgcn_wmma_f32_16x16x32_bf16(
          false, a0h, false, b_h, (short)0, acc0[j], false, false);
      acc0[j] = __builtin_amdgcn_wmma_f32_16x16x32_bf16(
          false, a0l, false, b_h, (short)0, acc0[j], false, false);
      acc0[j] = __builtin_amdgcn_wmma_f32_16x16x32_bf16(
          false, a0h, false, b_l, (short)0, acc0[j], false, false);
      acc1[j] = __builtin_amdgcn_wmma_f32_16x16x32_bf16(
          false, a1h, false, b_h, (short)0, acc1[j], false, false);
      acc1[j] = __builtin_amdgcn_wmma_f32_16x16x32_bf16(
          false, a1l, false, b_h, (short)0, acc1[j], false, false);
      acc1[j] = __builtin_amdgcn_wmma_f32_16x16x32_bf16(
          false, a1h, false, b_l, (short)0, acc1[j], false, false);
    }
    aoff += 512;
  }

  // C/D layout: lane<16 -> M=row0+v, N=lane; lane>=16 -> M=row0+8.. handled by
  // row0 = mt*16 + (lane>>4)*8, col = nt*16 + (lane&15).
  const int colb = lane & 15;
#pragma unroll
  for (int mm = 0; mm < 2; ++mm) {
    const int row0 = (mt0 + mm) * 16 + (lane >> 4) * 8;
#pragma unroll
    for (int j = 0; j < 4; ++j) {
      int col = (nt0 + j) * 16 + colb;
      float bv = bias[col];
      const v8f& a = mm ? acc1[j] : acc0[j];
#pragma unroll
      for (int v = 0; v < 8; ++v) {
        y[(size_t)(row0 + v) * OUT_F + col] = a[v] + bv;
      }
    }
  }
}

// ---------------------------------------------------------------------------
// K5: accum = log_dot_exp(accum_blocks, ldb). One thread per (b,d) 8x8 tile,
// streaming 256B in / 256B out; eyb/my are L2-resident (72 KB total).
// ---------------------------------------------------------------------------
__global__ __launch_bounds__(256) void logdotexp_kernel(
    const float* __restrict__ ab, const float* __restrict__ eyb,
    const float* __restrict__ my, float* __restrict__ out) {
  size_t t = (size_t)blockIdx.x * 256 + threadIdx.x;  // < 2048*256
  int d = (int)(t & 255);
  const float* lx = ab + t * 64;
  float* op = out + t * 64;

  float ey[64];
  const float4* ep = (const float4*)(eyb + (size_t)d * 64);
#pragma unroll
  for (int q = 0; q < 16; ++q) {
    float4 v = ep[q];
    ey[q * 4 + 0] = v.x; ey[q * 4 + 1] = v.y;
    ey[q * 4 + 2] = v.z; ey[q * 4 + 3] = v.w;
  }
  float myv[8];
#pragma unroll
  for (int j = 0; j < 8; ++j) myv[j] = my[d * 8 + j];

#pragma unroll
  for (int i = 0; i < 8; ++i) {
    float4 r0 = *(const float4*)(lx + i * 8);
    float4 r1 = *(const float4*)(lx + i * 8 + 4);
    float r[8] = {r0.x, r0.y, r0.z, r0.w, r1.x, r1.y, r1.z, r1.w};
    float mx = r[0];
#pragma unroll
    for (int k = 1; k < 8; ++k) mx = fmaxf(mx, r[k]);
    float e[8];
#pragma unroll
    for (int k = 0; k < 8; ++k) e[k] = __expf(r[k] - mx);
    float s[8] = {0.f, 0.f, 0.f, 0.f, 0.f, 0.f, 0.f, 0.f};
#pragma unroll
    for (int k = 0; k < 8; ++k)
#pragma unroll
      for (int j = 0; j < 8; ++j) s[j] = fmaf(e[k], ey[k * 8 + j], s[j]);
    float o[8];
#pragma unroll
    for (int j = 0; j < 8; ++j) o[j] = __logf(s[j]) + mx + myv[j];
    *(float4*)(op + i * 8)     = make_float4(o[0], o[1], o[2], o[3]);
    *(float4*)(op + i * 8 + 4) = make_float4(o[4], o[5], o[6], o[7]);
  }
}

// ---------------------------------------------------------------------------
extern "C" void kernel_launch(void* const* d_in, const int* in_sizes, int n_in,
                              void* d_out, int out_size, void* d_ws, size_t ws_size,
                              hipStream_t stream) {
  const float* x        = (const float*)d_in[0];
  const float* ab       = (const float*)d_in[1];
  const float* weight   = (const float*)d_in[2];
  const float* log_diag = (const float*)d_in[3];
  const float* bias     = (const float*)d_in[4];

  float* y         = (float*)d_out;
  float* accum_out = y + (size_t)BATCH * OUT_F;

  // Workspace layout (~33.7 MB total)
  char* ws = (char*)d_ws;
  unsigned short* Ah = (unsigned short*)(ws + 0);
  unsigned short* Al = (unsigned short*)(ws + 8388608);
  unsigned short* Bh = (unsigned short*)(ws + 16777216);
  unsigned short* Bl = (unsigned short*)(ws + 25165824);
  float* log_norm    = (float*)(ws + 33554432);          // 2048 f32
  float* eyb         = (float*)(ws + 33562624);          // 256*64 f32
  float* my          = (float*)(ws + 33628160);          // 256*8 f32

  prep_b_kernel<<<2048, 256, 0, stream>>>(weight, log_diag, log_norm, Bh, Bl);
  prep_a_kernel<<<2048, 256, 0, stream>>>(x, Ah, Al);
  prep_ldb_kernel<<<256, 64, 0, stream>>>(weight, log_diag, log_norm, eyb, my);
  gemm_bf16x3_kernel<<<dim3(32, 8), 256, 0, stream>>>(Ah, Al, Bh, Bl, bias, y);
  logdotexp_kernel<<<2048, 256, 0, stream>>>(ab, eyb, my, accum_out);
}
